// DMCNN_Encoder_argument_61134564491526
// MI455X (gfx1250) — compile-verified, matrix operand order checked by reading, not run on hardware
//
#include <hip/hip_runtime.h>
#include <hip/hip_bf16.h>
#include <math.h>

typedef __attribute__((ext_vector_type(16))) _Float16    v16h;
typedef __attribute__((ext_vector_type(8)))  float       v8f;
typedef __attribute__((ext_vector_type(8)))  unsigned int v8u;

#define SZV     2048
#define SENLEN  128
#define DWE     100
#define DPE     5
#define DEE     100
#define DCV     300
#define KLOC    10
#define CINV    210          // DWE + 2*DPE + DEE
#define KDIM    630          // CIN * 3
#define KTILES  20           // K padded to 640
#define MTILES  19           // M padded to 304
#define XSTRIDE 220          // f16 column stride (>= 220 so padded K reads land on zeros)
#define XCOLS   130          // 128 positions + 2 zero halo columns
#define MPAD    304

// ---------------------------------------------------------------------------
// Prep: conv_w (300,210,3) f32 -> f16 A-fragment tiles in d_ws, pre-swizzled
// into the CDNA5 16x32 f16 A-matrix lane layout (ISA 7.12.2): one contiguous
// v16h (32B) per lane per tile in the GEMM.
// ---------------------------------------------------------------------------
__global__ void dmcnn_prep_A(const float* __restrict__ conv_w,
                             _Float16* __restrict__ Atiles) {
    const int tile = blockIdx.x;           // mi*KTILES + kk
    const int mi   = tile / KTILES;
    const int kk   = tile - mi * KTILES;
    for (int e = threadIdx.x; e < 512; e += blockDim.x) {
        const int lane = e >> 4;
        const int q    = e & 15;
        const int r    = q >> 1;           // VGPR index 0..7
        const int h    = q & 1;            // low/high half of the K pair
        const int kc   = lane >> 4;        // upper lane half carries K+8
        const int kOff = ((r < 4) ? (2 * r + 8 * kc)
                                  : (16 + 2 * (r - 4) + 8 * kc)) + h;
        const int k    = kk * 32 + kOff;   // k = t*210 + c
        const int m    = mi * 16 + (lane & 15);
        float v = 0.0f;
        if (m < DCV && k < KDIM) {
            const int t = (k >= 420) ? 2 : ((k >= 210) ? 1 : 0);
            const int c = k - t * CINV;
            v = conv_w[(m * CINV + c) * 3 + t];   // conv_w[m][c][t]
        }
        Atiles[(tile << 9) + e] = (_Float16)v;
    }
}

// ---------------------------------------------------------------------------
// Main: one workgroup (8 wave32) per sample.
// ---------------------------------------------------------------------------
__global__ void __launch_bounds__(256)
dmcnn_encoder_kernel(const int* __restrict__ inp,
                     const int* __restrict__ pos1,
                     const int* __restrict__ pos2,
                     const int* __restrict__ loc,
                     const int* __restrict__ loc_mark,
                     const int* __restrict__ subtype,
                     const float* __restrict__ maskL,
                     const float* __restrict__ maskM,
                     const float* __restrict__ word_emb,
                     const float* __restrict__ pos_emb,
                     const float* __restrict__ event_emb,
                     const float* __restrict__ conv_b,
                     const _Float16* __restrict__ Atiles,
                     float* __restrict__ out) {
    __shared__ _Float16 xcol[XCOLS * XSTRIDE];   // column-major x, f16, zero halo
    __shared__ int toks[SENLEN], p1v[SENLEN], p2v[SENLEN];
    __shared__ unsigned char regc[SENLEN];       // 0=L 1=M 2=R per position
    __shared__ float biasLds[MPAD];
    __shared__ float pooledLds[3 * MPAD];

    const int s    = blockIdx.x;
    const int tid  = threadIdx.x;
    const int lane = tid & 31;
    const int wave = tid >> 5;

    // ---- Phase 0: zero LDS, stage per-position scalars --------------------
    {
        unsigned int* xz = (unsigned int*)xcol;
        for (int i = tid; i < (XCOLS * XSTRIDE) / 2; i += 256) xz[i] = 0u;
        for (int l = tid; l < SENLEN; l += 256) {
            toks[l] = inp[s * SENLEN + l];
            p1v[l]  = pos1[s * SENLEN + l];
            p2v[l]  = pos2[s * SENLEN + l];
            const float mL = maskL[s * SENLEN + l];
            const float mM = maskM[s * SENLEN + l];
            regc[l] = (mL > 0.5f) ? 0 : ((mM > 0.5f) ? 1 : 2);
        }
        for (int m = tid; m < MPAD; m += 256)
            biasLds[m] = (m < DCV) ? conv_b[m] : 0.0f;
        for (int i = tid; i < 3 * MPAD; i += 256)
            pooledLds[i] = 0.0f;   // masked positions contribute exactly 0 to the max
    }
    __syncthreads();

    // ---- Phase 1: gather embeddings -> xcol (f16) -------------------------
    {
        const int st = subtype[s];
        for (int idx = tid; idx < SENLEN * CINV; idx += 256) {
            const int l = idx / CINV;
            const int c = idx - l * CINV;
            float v;
            if (c < DWE) {
                const int tok = toks[l];
                v = (tok == 0) ? 0.0f : word_emb[tok * DWE + c];          // padding_idx
            } else if (c < DWE + DPE) {
                v = pos_emb[p1v[l] * DPE + (c - DWE)];
            } else if (c < DWE + 2 * DPE) {
                v = pos_emb[p2v[l] * DPE + (c - DWE - DPE)];
            } else {
                v = (st == 0) ? 0.0f : event_emb[st * DEE + (c - DWE - 2 * DPE)];
            }
            xcol[(l + 1) * XSTRIDE + c] = (_Float16)v;   // column l+1 (halo at 0,129)
        }
    }
    __syncthreads();

    // ---- Phase 2: implicit-im2col GEMM via WMMA ---------------------------
    // wave w owns N columns [16w, 16w+16).  B fragments (kk-dependent only)
    // are built once and stay register-resident across all 19 M-tiles.
    // B element (k,n) lives at f16 index 220*n + k + 10*t  (t = k/210).
    const int n  = wave * 16 + (lane & 15);
    const int kc = lane >> 4;                 // lane half carries K+8 / M+8
    const int nb = n * XSTRIDE;               // hoisted per-lane base

    v16h bfrag[KTILES];
    #pragma unroll
    for (int kk = 0; kk < KTILES; ++kk) {
        v8u bu;
        #pragma unroll
        for (int r = 0; r < 8; ++r) {
            const int kOff = (r < 4) ? (2 * r + 8 * kc)
                                     : (16 + 2 * (r - 4) + 8 * kc);
            const int k = kk * 32 + kOff;             // pair (k,k+1); never crosses t
            const int t = (k >= 420) ? 2 : ((k >= 210) ? 1 : 0);
            bu[r] = *(const unsigned int*)&xcol[nb + k + 10 * t];
        }
        bfrag[kk] = __builtin_bit_cast(v16h, bu);
    }

    for (int mi = 0; mi < MTILES; ++mi) {
        const _Float16* ap = Atiles + (((mi * KTILES) << 9) + (lane << 4));
        v8f acc0 = {}, acc1 = {};
        v16h acur = *(const v16h*)ap;             // A tile kk=0
        #pragma unroll
        for (int kk = 0; kk < KTILES; ++kk) {
            v16h anxt = acur;
            if (kk + 1 < KTILES)                  // load next A tile while wmma runs
                anxt = *(const v16h*)(ap + ((kk + 1) << 9));
            if ((kk & 1) == 0)                    // two independent acc chains (ILP)
                acc0 = __builtin_amdgcn_wmma_f32_16x16x32_f16(
                           false, acur, false, bfrag[kk], (short)0, acc0, false, false);
            else
                acc1 = __builtin_amdgcn_wmma_f32_16x16x32_f16(
                           false, acur, false, bfrag[kk], (short)0, acc1, false, false);
            acur = anxt;
        }
        // epilogue: + bias, region-wise max-pool (padded rows are 0 and land in
        // pooled slots 300..303 which are never read -> no divergence needed)
        const int reg = regc[n];
        float* pr = &pooledLds[reg * MPAD + mi * 16 + 8 * kc];
        #pragma unroll
        for (int r = 0; r < 8; ++r)
            atomicMax(&pr[r], acc0[r] + acc1[r] + biasLds[mi * 16 + r + 8 * kc]);
    }

    // ---- Phase 3: loc features (600 floats), tanh, write ------------------
    {
        const int mark = loc_mark[s];
        float* outs = out + (size_t)s * 1500;
        for (int idx = tid; idx < 600; idx += 256) {
            float v;
            if (idx < 400) {
                const int j = idx / DWE, c = idx - j * DWE;
                const int tok = loc[s * KLOC + j];
                v = (tok == 0) ? 0.0f : word_emb[tok * DWE + c];
            } else if (idx < 500) {
                const int c = idx - 400;
                float acc = 0.0f;
                for (int j = 4; j < 4 + mark; ++j) {
                    const int tok = loc[s * KLOC + j];
                    acc += (tok == 0) ? 0.0f : word_emb[tok * DWE + c];
                }
                v = acc / (float)mark;
            } else {
                const int c = idx - 500;
                const int tok = loc[s * KLOC + 4 + mark];
                v = (tok == 0) ? 0.0f : word_emb[tok * DWE + c];
            }
            outs[900 + idx] = tanhf(v);
        }
    }
    __syncthreads();

    // ---- Phase 4: write tanh(pooled) --------------------------------------
    {
        float* outs = out + (size_t)s * 1500;
        for (int idx = tid; idx < 900; idx += 256) {
            const int reg = idx / DCV;
            const int m   = idx - reg * DCV;
            outs[idx] = tanhf(pooledLds[reg * MPAD + m]);
        }
    }
}

// ---------------------------------------------------------------------------
extern "C" void kernel_launch(void* const* d_in, const int* in_sizes, int n_in,
                              void* d_out, int out_size, void* d_ws, size_t ws_size,
                              hipStream_t stream) {
    const int*   inp       = (const int*)  d_in[0];
    const int*   pos1      = (const int*)  d_in[1];
    const int*   pos2      = (const int*)  d_in[2];
    const int*   loc       = (const int*)  d_in[3];
    const int*   loc_mark  = (const int*)  d_in[4];
    const int*   subtype   = (const int*)  d_in[5];
    const float* maskL     = (const float*)d_in[6];
    const float* maskM     = (const float*)d_in[7];
    // maskR (d_in[8]) implied by !L && !M
    const float* word_emb  = (const float*)d_in[9];
    const float* pos_emb   = (const float*)d_in[10];
    const float* event_emb = (const float*)d_in[11];
    const float* conv_w    = (const float*)d_in[12];
    const float* conv_b    = (const float*)d_in[13];
    float*       out       = (float*)d_out;

    _Float16* Atiles = (_Float16*)d_ws;   // MTILES*KTILES*512 f16 = 380 KB

    dmcnn_prep_A<<<MTILES * KTILES, 256, 0, stream>>>(conv_w, Atiles);
    dmcnn_encoder_kernel<<<SZV, 256, 0, stream>>>(
        inp, pos1, pos2, loc, loc_mark, subtype, maskL, maskM,
        word_emb, pos_emb, event_emb, conv_b, Atiles, out);
}
